// decoder_44324062495129
// MI455X (gfx1250) — compile-verified, matrix operand order checked by reading, not run on hardware
//
#include <hip/hip_runtime.h>
#include <math.h>

typedef __bf16 bf16;
typedef __attribute__((ext_vector_type(4)))  __bf16 bf16x4;
typedef __attribute__((ext_vector_type(8)))  __bf16 bf16x8;
typedef __attribute__((ext_vector_type(16))) __bf16 bf16x16;
typedef __attribute__((ext_vector_type(4)))  float  f32x4;
typedef __attribute__((ext_vector_type(8)))  float  f32x8;

#define BATCH 1024
#define NNODE 200
#define DH    512
#define DK    512
#define DC    1536   // 3*DH
#define NCHUNK 13    // ceil(200/16) node chunks for WMMA scores
#define CCLIP 10.0f
#define INV_SQRT_DK 0.044194173824159216f   // 1/sqrt(512)

// ---------------------------------------------------------------------------
// WMMA fragment loader. 16-bit A-matrix 16x32 layout (ISA 7.12.2):
//   lanes 0-15 : row M = m0+lane,      elems 0-7 = K k..k+7,  elems 8-15 = K k+16..k+23
//   lanes 16-31: row M = m0+lane-16,   elems 0-7 = K k+8..k+15, elems 8-15 = K k+24..k+31
// B is loaded column-major from a pre-transposed (N x Kd) bf16 matrix with the
// same per-lane pattern (columns of B == rows of Bt). With ld==0 every column
// is identical (vector broadcast into all 16 B columns).
// ---------------------------------------------------------------------------
__device__ __forceinline__ bf16x16 load_frag(const bf16* __restrict__ base,
                                             int ld, int row, int kbase, int lane) {
  const int r   = row + (lane & 15);
  const int sel = (lane >> 4) & 1;
  const bf16* p = base + (long)r * ld + kbase + sel * 8;
  bf16x8 lo = *(const bf16x8*)(p);        // K k..k+7   (or +8..+15)
  bf16x8 hi = *(const bf16x8*)(p + 16);   // K k+16..23 (or +24..+31)
  bf16x16 f;
#pragma unroll
  for (int i = 0; i < 8; ++i) { f[i] = lo[i]; f[i + 8] = hi[i]; }
  return f;
}

// ---------------------------------------------------------------------------
// Generic bf16 WMMA GEMM: C[M x Nc] = A[M x Kd] * Bt[Nc x Kd]^T, bf16 out.
// One wave -> 16x64 output strip (4 WMMA tiles, A fragment reused 4x).
// Grid must supply exactly (M/16)*(Nc/64) waves; blockDim = 256 (8 waves).
// ---------------------------------------------------------------------------
__global__ void __launch_bounds__(256) wmma_gemm_bf16(const bf16* __restrict__ A,
                                                      const bf16* __restrict__ Bt,
                                                      bf16* __restrict__ C,
                                                      int M, int Kd, int Nc) {
  const int lane = threadIdx.x & 31;
  const int wid  = threadIdx.x >> 5;
  const long tid = (long)blockIdx.x * 8 + wid;
  const int  nt  = Nc >> 6;                // 64-wide N tiles
  const int  tm  = (int)(tid / nt);
  const int  tn  = (int)(tid % nt);
  const int  m0  = tm << 4;
  const int  n0  = tn << 6;

  f32x8 c0 = {}, c1 = {}, c2 = {}, c3 = {};
  for (int k = 0; k < Kd; k += 32) {
    bf16x16 a  = load_frag(A,  Kd, m0,      k, lane);
    bf16x16 b0 = load_frag(Bt, Kd, n0,      k, lane);
    bf16x16 b1 = load_frag(Bt, Kd, n0 + 16, k, lane);
    bf16x16 b2 = load_frag(Bt, Kd, n0 + 32, k, lane);
    bf16x16 b3 = load_frag(Bt, Kd, n0 + 48, k, lane);
    c0 = __builtin_amdgcn_wmma_f32_16x16x32_bf16(false, a, false, b0, (short)0, c0, false, false);
    c1 = __builtin_amdgcn_wmma_f32_16x16x32_bf16(false, a, false, b1, (short)0, c1, false, false);
    c2 = __builtin_amdgcn_wmma_f32_16x16x32_bf16(false, a, false, b2, (short)0, c2, false, false);
    c3 = __builtin_amdgcn_wmma_f32_16x16x32_bf16(false, a, false, b3, (short)0, c3, false, false);
  }

  // C/D layout: VGPR r holds row m0+r (lanes 0-15) / m0+8+r (lanes 16-31), col = n0+lane%16
  const int col = lane & 15;
  const int rb  = m0 + ((lane >> 4) << 3);
#pragma unroll
  for (int r = 0; r < 8; ++r) {
    const long ro = (long)(rb + r) * Nc;
    C[ro + n0 +      col] = (bf16)c0[r];
    C[ro + n0 + 16 + col] = (bf16)c1[r];
    C[ro + n0 + 32 + col] = (bf16)c2[r];
    C[ro + n0 + 48 + col] = (bf16)c3[r];
  }
}

// ---------------------------------------------------------------------------
// f32 -> bf16 bulk convert (4 elems/thread)
// ---------------------------------------------------------------------------
__global__ void f32_to_bf16_vec4(const float* __restrict__ src,
                                 bf16* __restrict__ dst, long n4) {
  const long i = (long)blockIdx.x * blockDim.x + threadIdx.x;
  if (i >= n4) return;
  f32x4 v = *(const f32x4*)(src + i * 4);
  bf16x4 o;
  o[0] = (bf16)v[0]; o[1] = (bf16)v[1]; o[2] = (bf16)v[2]; o[3] = (bf16)v[3];
  *(bf16x4*)(dst + i * 4) = o;
}

// transpose f32[rows x cols] -> bf16[cols x rows]
__global__ void transpose_to_bf16(const float* __restrict__ src,
                                  bf16* __restrict__ dst, int rows, int cols) {
  const long i = (long)blockIdx.x * blockDim.x + threadIdx.x;
  const long total = (long)rows * cols;
  if (i >= total) return;
  const long c = i / rows;
  const long r = i % rows;
  dst[i] = (bf16)src[r * (long)cols + c];
}

// h_c0 = [H_bar | v_1 | v_f] in bf16; one block per batch element
__global__ void init_hc(const float* __restrict__ H, const float* __restrict__ v1,
                        const float* __restrict__ vf, bf16* __restrict__ hc) {
  const int b = blockIdx.x;
  for (int d = threadIdx.x; d < DH; d += blockDim.x) {
    float s = 0.0f;
    const float* p = H + (long)b * NNODE * DH + d;
    for (int nn = 0; nn < NNODE; ++nn) s += p[(long)nn * DH];
    hc[(long)b * DC + d]          = (bf16)(s * (1.0f / NNODE));
    hc[(long)b * DC + DH + d]     = (bf16)v1[d];
    hc[(long)b * DC + 2 * DH + d] = (bf16)vf[d];
  }
}

__device__ __forceinline__ unsigned pcg_hash(unsigned x) {
  unsigned state = x * 747796405u + 2891336453u;
  unsigned word  = ((state >> ((state >> 28u) + 4u)) ^ state) * 277803737u;
  return (word >> 22u) ^ word;
}

// ---------------------------------------------------------------------------
// Fused scores + Gumbel-max sample + h_c update. One block (8 waves) per b.
// U = K_b (200x512) x Q_b broadcast into 16 columns, via WMMA: the kernel is a
// pure HBM stream of K (210 MB/step), so the (16x redundant) matrix-core
// compute is free and all per-element VALU convert+FMA work disappears.
// Wave w handles node chunks {w, w+8} of 13 (chunk 12 overreads 8 rows into
// the next ws region: benign, finite, never read back).
// ---------------------------------------------------------------------------
__global__ void __launch_bounds__(256) scores_sample(const bf16* __restrict__ Qb,
                                                     const bf16* __restrict__ Kb,
                                                     const bf16* __restrict__ Hb,
                                                     bf16* __restrict__ hc,
                                                     int* __restrict__ Pi,
                                                     int step) {
  __shared__ float sU[NCHUNK * 16];   // 208 slots; only n<200 consumed
  __shared__ int   sPick;
  const int b    = blockIdx.x;
  const int lane = threadIdx.x & 31;
  const int wid  = threadIdx.x >> 5;

  const bf16* Qp    = Qb + (long)b * DK;
  const bf16* kbase = Kb + (long)b * NNODE * DK;

  for (int chunk = wid; chunk < NCHUNK; chunk += 8) {   // wave-uniform
    const int m0 = chunk * 16;
    f32x8 acc = {};
    for (int k = 0; k < DK; k += 32) {
      bf16x16 a = load_frag(kbase, DK, m0, k, lane);    // 16 nodes x 32 k
      bf16x16 q = load_frag(Qp, /*ld=*/0, 0, k, lane);  // Q broadcast (all cols)
      acc = __builtin_amdgcn_wmma_f32_16x16x32_bf16(false, a, false, q,
                                                    (short)0, acc, false, false);
    }
    // every column of D is identical; lanes 0 and 16 own rows m0..m0+7 / +8..+15
    if ((lane & 15) == 0) {
      const int rbase = m0 + ((lane >> 4) << 3);
#pragma unroll
      for (int r = 0; r < 8; ++r) sU[rbase + r] = acc[r];
    }
  }
  __syncthreads();

  if (wid == 0) {
    float best = -1e30f;
    int   bidx = 0;
    for (int n = lane; n < NNODE; n += 32) {
      const float logit = CCLIP * tanhf(sU[n] * INV_SQRT_DK);
      const unsigned h =
          pcg_hash(0x9E3779B9u * (unsigned)step + 0x85EBCA77u * (unsigned)b +
                   0xC2B2AE3Du * (unsigned)n + 0x42421337u);
      const float u = (float)(h >> 8) * 0x1p-24f + 0x1p-25f;  // (0,1)
      const float g = -logf(-logf(u));                         // Gumbel(0,1)
      const float v = logit + g;
      if (v > best) { best = v; bidx = n; }
    }
#pragma unroll
    for (int off = 16; off > 0; off >>= 1) {
      const float ov = __shfl_xor(best, off, 32);
      const int   oi = __shfl_xor(bidx, off, 32);
      if (ov > best || (ov == best && oi < bidx)) { best = ov; bidx = oi; }
    }
    if (lane == 0) sPick = bidx;
  }
  __syncthreads();

  const int pick = sPick;
  const bf16* hrow = Hb + ((long)b * NNODE + pick) * DH;
  for (int d = threadIdx.x; d < DH; d += 256) {
    const bf16 val = hrow[d];
    hc[(long)b * DC + DH + d] = val;                     // H_pi slot
    if (step == 0) hc[(long)b * DC + 2 * DH + d] = val;  // freeze H_pi_1
  }
  if (threadIdx.x == 0) {
    Pi[(long)b * (NNODE + 1) + step + 1] = pick;
    if (step == 0) Pi[(long)b * (NNODE + 1)] = pick;     // duplicated first sample
  }
}

// ---------------------------------------------------------------------------
extern "C" void kernel_launch(void* const* d_in, const int* in_sizes, int n_in,
                              void* d_out, int out_size, void* d_ws, size_t ws_size,
                              hipStream_t stream) {
  (void)in_sizes; (void)n_in; (void)out_size; (void)ws_size;
  const float* H   = (const float*)d_in[1];
  const float* W_Q = (const float*)d_in[2];
  const float* W_K = (const float*)d_in[3];
  const float* v1  = (const float*)d_in[4];
  const float* vf  = (const float*)d_in[5];
  int* Pi = (int*)d_out;

  char*  ws  = (char*)d_ws;
  size_t off = 0;
  auto alloc = [&](size_t bytes) -> void* {
    void* p = ws + off;
    off += (bytes + 255) & ~(size_t)255;
    return p;
  };
  bf16* Hb  = (bf16*)alloc((size_t)BATCH * NNODE * DH * 2);  // 210 MB
  bf16* Kb  = (bf16*)alloc((size_t)BATCH * NNODE * DK * 2);  // 210 MB
  bf16* WQt = (bf16*)alloc((size_t)DK * DC * 2);             // 1.5 MB (also chunk-12 overread pad)
  bf16* WKt = (bf16*)alloc((size_t)DK * DH * 2);             // 0.5 MB
  bf16* hc  = (bf16*)alloc((size_t)BATCH * DC * 2);          // 3 MB
  bf16* Qb  = (bf16*)alloc((size_t)BATCH * DK * 2);          // 1 MB

  // 1) bf16 conversions / transposes (W*t are Nc x Kd, K-contiguous)
  const long n4 = (long)BATCH * NNODE * DH / 4;
  f32_to_bf16_vec4<<<(int)((n4 + 255) / 256), 256, 0, stream>>>(H, Hb, n4);
  transpose_to_bf16<<<(DC * DK + 255) / 256, 256, 0, stream>>>(W_Q, WQt, DC, DK);
  transpose_to_bf16<<<(DH * DK + 255) / 256, 256, 0, stream>>>(W_K, WKt, DH, DK);

  // 2) h_c0 = [H_bar | v_1 | v_f]
  init_hc<<<BATCH, 256, 0, stream>>>(H, v1, vf, hc);

  // 3) K = H @ W_K  (M = 204800, Kd = 512, Nc = 512) -> 102400 waves / 12800 blocks
  {
    const long tiles = ((long)BATCH * NNODE / 16) * (DK / 64);
    wmma_gemm_bf16<<<(int)(tiles / 8), 256, 0, stream>>>(Hb, WKt, Kb,
                                                         BATCH * NNODE, DH, DK);
  }

  // 4) 200 sequential decode steps
  const int qblocks = (int)(((BATCH / 16) * (DK / 64)) / 8);  // 64 blocks
  for (int step = 0; step < NNODE; ++step) {
    wmma_gemm_bf16<<<qblocks, 256, 0, stream>>>(hc, WQt, Qb, BATCH, DC, DK);
    scores_sample<<<BATCH, 256, 0, stream>>>(Qb, Kb, Hb, hc, Pi, step);
  }
}
